// MultiheadAttention_24653112279189
// MI455X (gfx1250) — matrix operand order repaired and checked by reading, not compiled, not run on hardware
//
#include <hip/hip_runtime.h>
#include <cstdint>

// MultiheadAttention + LoRA for MI455X (gfx1250, wave32, WMMA bf16 16x16x32).
// Pipeline (all on `stream`):
//   k_merge : Weff[h][p][d][e] = W_p[h,d,e] + sum_r A[h,p,d,r]*B[h,p,e,r]  (bf16)
//   k_tcvt  : q,k,v [B,E,S] f32 -> qT,kT,vT [B,S,E] bf16
//   k_proj  : QT,KT [B,H,S,E] bf16 (transposed), V [B,H,E,S] bf16
//   k_attn  : flash attention (softmax over s per column j, ref axis=2)
//             Kt j-tile loaded via Tensor Data Mover (TDM) when available
//             -> AOT [B,H,S,E] bf16
//   k_out   : out[b,d,s] = sum_h sum_e Wo_eff[h,d,e]*AOT[b,h,s,e]  (f32)

#if __has_builtin(__builtin_amdgcn_tensor_load_to_lds) && \
    __has_builtin(__builtin_amdgcn_s_wait_tensorcnt)
#define ATHENA_USE_TDM 1
#else
#define ATHENA_USE_TDM 0
#endif

namespace {

constexpr int NB = 2, NE = 128, NS = 2048, NH = 8, NR = 8;

typedef __attribute__((ext_vector_type(16))) __bf16 bf16x16;
typedef __attribute__((ext_vector_type(8)))  __bf16 bf16x8;
typedef __attribute__((ext_vector_type(8)))  float  f32x8;
#if ATHENA_USE_TDM
typedef __attribute__((ext_vector_type(4))) unsigned int u32x4;
typedef __attribute__((ext_vector_type(8))) int          i32x8;
typedef __attribute__((ext_vector_type(4))) int          i32x4;
#endif

// Workspace layout (bytes)
constexpr size_t WEFF_B = (size_t)NH * 4 * NE * NE * 2;   // 1 MB
constexpr size_t XT_B   = (size_t)NB * NS * NE * 2;       // 1 MB each (qT/kT/vT)
constexpr size_t PRJ_B  = (size_t)NB * NH * NS * NE * 2;  // 8 MB each (QT/KT/V/AOT)
constexpr size_t OFF_WEFF = 0;
constexpr size_t OFF_QTI  = OFF_WEFF + WEFF_B;
constexpr size_t OFF_KTI  = OFF_QTI + XT_B;
constexpr size_t OFF_VTI  = OFF_KTI + XT_B;
constexpr size_t OFF_QT   = OFF_VTI + XT_B;
constexpr size_t OFF_KT   = OFF_QT + PRJ_B;
constexpr size_t OFF_V    = OFF_KT + PRJ_B;
constexpr size_t OFF_AOT  = OFF_V  + PRJ_B;
constexpr size_t WS_NEED  = OFF_AOT + PRJ_B;              // 36 MB total

// k_attn dynamic LDS partition (elements of __bf16), rows padded by 8 (16B)
constexpr int KT_ROWS = 128, QT_ROWS = 64, LROW = 136;    // [row][136]
constexpr int V_ROWS = 128, VROW = 72;                    // [e][72]
constexpr int SM_KT = 0;
constexpr int SM_QT = SM_KT + KT_ROWS * LROW;
constexpr int SM_V  = SM_QT + QT_ROWS * LROW;
constexpr size_t SMEM_ATTN_B = (size_t)(SM_V + V_ROWS * VROW) * 2;  // 70656 B

__device__ __forceinline__ bf16x16 ld2x8(const __bf16* p0, const __bf16* p1) {
  bf16x8 lo = *(const bf16x8*)p0;   // 16B load
  bf16x8 hi = *(const bf16x8*)p1;   // 16B load
  return __builtin_shufflevector(lo, hi, 0,1,2,3,4,5,6,7,8,9,10,11,12,13,14,15);
}

__device__ __forceinline__ f32x8 wmma_bf16(bf16x16 a, bf16x16 b, f32x8 c) {
  return __builtin_amdgcn_wmma_f32_16x16x32_bf16(false, a, false, b, (short)0, c,
                                                 false, false);
}

} // namespace

// ---------------- k_merge : LoRA merge + convert to bf16 ----------------
__global__ void k_merge(const float* __restrict__ Wq, const float* __restrict__ Wk,
                        const float* __restrict__ Wv, const float* __restrict__ Wo,
                        const float* __restrict__ lA, const float* __restrict__ lB,
                        __bf16* __restrict__ weff) {
  int idx = blockIdx.x * blockDim.x + threadIdx.x;       // over H*4*E*E
  if (idx >= NH * 4 * NE * NE) return;
  int e = idx % NE;
  int d = (idx / NE) % NE;
  int p = (idx / (NE * NE)) % 4;
  int h = idx / (4 * NE * NE);
  const float* W = (p == 0) ? Wq : (p == 1) ? Wk : (p == 2) ? Wv : Wo;
  float s = W[(h * NE + d) * NE + e];
  const float* a = lA + ((size_t)(h * 4 + p) * NE + d) * NR;
  const float* b = lB + ((size_t)(h * 4 + p) * NE + e) * NR;
#pragma unroll
  for (int r = 0; r < NR; ++r) s += a[r] * b[r];
  weff[idx] = (__bf16)s;   // layout [h][p][d][e]
}

// ---------------- k_tcvt : transpose + convert inputs ----------------
__global__ void k_tcvt(const float* __restrict__ q, const float* __restrict__ k,
                       const float* __restrict__ v, __bf16* __restrict__ qT,
                       __bf16* __restrict__ kT, __bf16* __restrict__ vT) {
  int idx = blockIdx.x * blockDim.x + threadIdx.x;       // over B*S*E
  if (idx >= NB * NS * NE) return;
  int z = blockIdx.z;
  const float* x = (z == 0) ? q : (z == 1) ? k : v;
  __bf16* xt = (z == 0) ? qT : (z == 1) ? kT : vT;
  int e = idx % NE;
  int s = (idx / NE) % NS;
  int b = idx / (NE * NS);
  xt[idx] = (__bf16)x[((size_t)b * NE + e) * NS + s];    // [b][s][e] <- [b][e][s]
}

// ---------------- k_proj : Q/K/V projections ----------------
// grid (S/64, B*H, 3 projections), 128 threads (4 waves).
__global__ __launch_bounds__(128) void k_proj(const __bf16* __restrict__ weff,
                                              const __bf16* __restrict__ qT,
                                              const __bf16* __restrict__ kT,
                                              const __bf16* __restrict__ vT,
                                              __bf16* __restrict__ QT,
                                              __bf16* __restrict__ KT,
                                              __bf16* __restrict__ V) {
  const int p    = blockIdx.z;
  const int bh   = blockIdx.y;
  const int b    = bh / NH;
  const int h    = bh % NH;
  const int s0   = blockIdx.x * 64;
  const int wave = threadIdx.x >> 5;
  const int lane = threadIdx.x & 31;
  const int l16  = lane & 15;
  const int half = lane >> 4;
  const int scol = s0 + wave * 16 + l16;                 // this lane's N column

  const __bf16* xT = ((p == 0) ? qT : (p == 1) ? kT : vT) + (size_t)b * NS * NE;
  const __bf16* A  = weff + (size_t)(h * 4 + p) * NE * NE;

  f32x8 acc[8] = {};
#pragma unroll
  for (int kk = 0; kk < 4; ++kk) {
    const int e0 = kk * 32;
    const __bf16* brow = xT + (size_t)scol * NE + e0 + half * 8;
    bf16x16 bv = ld2x8(brow, brow + 16);                 // B: col s, split K chunks
#pragma unroll
    for (int m = 0; m < 8; ++m) {
      const __bf16* arow = A + (size_t)(m * 16 + l16) * NE + e0 + half * 8;
      bf16x16 av = ld2x8(arow, arow + 16);               // A: row d, split K chunks
      acc[m] = wmma_bf16(av, bv, acc[m]);
    }
  }
  if (p < 2) {
    // store transposed: QT/KT[b][h][s][d]
    __bf16* dst = ((p == 0) ? QT : KT) + ((size_t)bh * NS + scol) * NE;
#pragma unroll
    for (int m = 0; m < 8; ++m) {
      bf16x8 pk;
#pragma unroll
      for (int i = 0; i < 8; ++i) pk[i] = (__bf16)acc[m][i];
      *(bf16x8*)(dst + m * 16 + half * 8) = pk;          // 16B store
    }
  } else {
    // store V natural [b][h][e][s]
    __bf16* dst = V + (size_t)bh * NE * NS;
#pragma unroll
    for (int m = 0; m < 8; ++m)
#pragma unroll
      for (int i = 0; i < 8; ++i)
        dst[(size_t)(m * 16 + half * 8 + i) * NS + scol] = (__bf16)acc[m][i];
  }
}

// ---------------- k_attn : fused flash attention ----------------
// softmax over s per column j:  out[:,j] = V * softmax_s(Q^T K[:,j])
// grid (S/128 j-tiles, B*H), 128 threads (4 waves). Wave owns 32 j-columns
// (2 groups of 16): A/B operands reused across groups -> 64 WMMA per s-iter.
__global__ __launch_bounds__(128) void k_attn(const __bf16* __restrict__ QT,
                                              const __bf16* __restrict__ KT,
                                              const __bf16* __restrict__ V,
                                              __bf16* __restrict__ AOT) {
  extern __shared__ __attribute__((aligned(16))) __bf16 smem[];
  __bf16* kt_l = smem + SM_KT;   // [128][136]
  __bf16* qt_l = smem + SM_QT;   // [64][136]
  __bf16* v_l  = smem + SM_V;    // [128][72]

  const int bh    = blockIdx.y;
  const int j0    = blockIdx.x * 128;
  const int tid   = threadIdx.x;
  const int wave  = tid >> 5;
  const int lane  = tid & 31;
  const int l16   = lane & 15;
  const int half  = lane >> 4;
  const int jbase = wave * 32;

  const __bf16* QTg = QT + (size_t)bh * NS * NE;
  const __bf16* KTg = KT + (size_t)bh * NS * NE;
  const __bf16* Vg  = V  + (size_t)bh * NE * NS;

#if ATHENA_USE_TDM
  // Resident K^T j-tile via Tensor Data Mover: 128x128 bf16 tile, LDS rows
  // padded by 4 DWORDs (16B) after every 64 DWORDs (one 256B row).
  if (wave == 0) {
    const unsigned long long ga =
        (unsigned long long)(uintptr_t)(KTg + (size_t)j0 * NE);
    u32x4 g0;
    g0[0] = 1u;                                    // count=1 (valid), load mode
    g0[1] = (unsigned)(uintptr_t)kt_l;             // lds_addr (low32 = LDS offset)
    g0[2] = (unsigned)(ga & 0xffffffffu);          // global_addr[31:0]
    g0[3] = (unsigned)((ga >> 32) & 0x01ffffffu) | (2u << 30);  // addr[56:32],type=2
    i32x8 g1;
    g1[0] = (1 << 16)     // data_size = 2 bytes
          | (1 << 20)     // pad_enable
          | (5 << 22)     // pad_interval: every 64 DWORDs
          | (3 << 25);    // pad_amount: 4 DWORDs
    g1[1] = (int)(128u << 16);   // tensor_dim0 = 128 (bits 79:48)
    g1[2] = (int)(128u << 16);   // tensor_dim1 = 128 (bits 111:80)
    g1[3] = (int)(128u << 16);   // tile_dim0 = 128 (bits 127:112)
    g1[4] = 128;                 // tile_dim1 = 128 rows
    g1[5] = 128;                 // tensor_dim0_stride = 128 elements
    g1[6] = 0;
    g1[7] = 0;
    i32x4 gz4 = {0, 0, 0, 0};                      // groups 2/3: NULL (2D tensor)
    i32x8 gz8 = {0, 0, 0, 0, 0, 0, 0, 0};
    __builtin_amdgcn_tensor_load_to_lds(g0, g1, gz4, gz4, gz8, 0);
    __builtin_amdgcn_s_wait_tensorcnt((short)0);
  }
#else
  for (int c = tid; c < 128 * 16; c += 128) {
    int r = c >> 4, c8 = c & 15;
    *(uint4*)&kt_l[r * LROW + c8 * 8] =
        *(const uint4*)&KTg[(size_t)(j0 + r) * NE + c8 * 8];
  }
#endif

  float rmax[2] = {-3.0e38f, -3.0e38f};
  float lsum[2] = {0.f, 0.f};
  f32x8 acc[2][8] = {};                            // accT[j 16 rows][e 128] x2

  for (int s0 = 0; s0 < NS; s0 += 64) {
    __syncthreads();                               // prior reads done
    for (int c = tid; c < 64 * 16; c += 128) {     // Q^T s-tile
      int r = c >> 4, c8 = c & 15;
      *(uint4*)&qt_l[r * LROW + c8 * 8] =
          *(const uint4*)&QTg[(size_t)(s0 + r) * NE + c8 * 8];
    }
    for (int c = tid; c < 128 * 8; c += 128) {     // V tile [e][64]
      int r = c >> 3, c8 = c & 7;
      *(uint4*)&v_l[r * VROW + c8 * 8] =
          *(const uint4*)&Vg[(size_t)r * NS + s0 + c8 * 8];
    }
    __syncthreads();

    // logits: C[s=64][j=16] x 2 j-groups; A (Q^T) reused across groups
    f32x8 lg[2][4] = {};
#pragma unroll
    for (int kk = 0; kk < 4; ++kk) {
      const int d0 = kk * 32;
      bf16x16 bvj[2];
#pragma unroll
      for (int jg = 0; jg < 2; ++jg) {
        const __bf16* bp = &kt_l[(jbase + jg * 16 + l16) * LROW + d0 + half * 8];
        bvj[jg] = ld2x8(bp, bp + 16);              // B: K^T row j, K=d
      }
#pragma unroll
      for (int t = 0; t < 4; ++t) {
        const __bf16* ap = &qt_l[(t * 16 + l16) * LROW + d0 + half * 8];
        bf16x16 av = ld2x8(ap, ap + 16);           // A: Q^T row s, K=d
#pragma unroll
        for (int jg = 0; jg < 2; ++jg)
          lg[jg][t] = wmma_bf16(av, bvj[jg], lg[jg][t]);
      }
    }

    // online softmax over s (per-lane stats for column j = l16, per group)
    float alpha[2];
#pragma unroll
    for (int jg = 0; jg < 2; ++jg) {
      float cmax = -3.0e38f;
#pragma unroll
      for (int t = 0; t < 4; ++t)
#pragma unroll
        for (int i = 0; i < 8; ++i) cmax = fmaxf(cmax, lg[jg][t][i]);
      cmax = fmaxf(cmax, __shfl_xor(cmax, 16, 32));
      float mnew = fmaxf(rmax[jg], cmax);
      alpha[jg]  = __expf(rmax[jg] - mnew);
      float rs = 0.f;
#pragma unroll
      for (int t = 0; t < 4; ++t)
#pragma unroll
        for (int i = 0; i < 8; ++i) {
          float pe = __expf(lg[jg][t][i] - mnew);
          lg[jg][t][i] = pe;                       // P in C-layout
          rs += pe;
        }
      rs += __shfl_xor(rs, 16, 32);
      lsum[jg] = lsum[jg] * alpha[jg] + rs;
      rmax[jg] = mnew;
      float sc[8];
#pragma unroll
      for (int i = 0; i < 8; ++i) sc[i] = __shfl(alpha[jg], half * 8 + i, 32);
#pragma unroll
      for (int n = 0; n < 8; ++n)
#pragma unroll
        for (int i = 0; i < 8; ++i) acc[jg][n][i] *= sc[i];
    }

    // PV: accT[j][e] += P^T x V^T; stacked C-tiles == A-layout of P^T;
    // V B-operand reused across both j-groups.
#pragma unroll
    for (int k2 = 0; k2 < 2; ++k2) {
      bf16x16 avp[2];
#pragma unroll
      for (int jg = 0; jg < 2; ++jg)
#pragma unroll
        for (int c = 0; c < 8; ++c) {
          avp[jg][c]     = (__bf16)lg[jg][2 * k2][c];
          avp[jg][8 + c] = (__bf16)lg[jg][2 * k2 + 1][c];
        }
#pragma unroll
      for (int n = 0; n < 8; ++n) {
        const __bf16* bp = &v_l[(n * 16 + l16) * VROW + k2 * 32 + half * 8];
        bf16x16 bv = ld2x8(bp, bp + 16);           // B: V row e, K=s
#pragma unroll
        for (int jg = 0; jg < 2; ++jg)
          acc[jg][n] = wmma_bf16(avp[jg], bv, acc[jg][n]);
      }
    }
  }

  // epilogue: divide by l, store AOT[b][h][j][e]
  __bf16* dst = AOT + (size_t)bh * NS * NE;
#pragma unroll
  for (int jg = 0; jg < 2; ++jg) {
    float inv = 1.f / lsum[jg];
    float wi[8];
#pragma unroll
    for (int i = 0; i < 8; ++i) wi[i] = __shfl(inv, half * 8 + i, 32);
#pragma unroll
    for (int n = 0; n < 8; ++n)
#pragma unroll
      for (int i = 0; i < 8; ++i) {
        int j = j0 + jbase + jg * 16 + half * 8 + i;
        int e = n * 16 + l16;
        dst[(size_t)j * NE + e] = (__bf16)(acc[jg][n][i] * wi[i]);
      }
  }
}

// ---------------- k_out : output projection, sum over heads ----------------
// out^T[s,d] = sum_h AOT_h[s,:] x Wo_eff[h,d,:]^T ; grid (S/64, B), 4 waves
__global__ __launch_bounds__(128) void k_out(const __bf16* __restrict__ AOT,
                                             const __bf16* __restrict__ weff,
                                             float* __restrict__ out) {
  const int b    = blockIdx.y;
  const int s0   = blockIdx.x * 64;
  const int wave = threadIdx.x >> 5;
  const int lane = threadIdx.x & 31;
  const int l16  = lane & 15;
  const int half = lane >> 4;
  const int srow = s0 + wave * 16 + l16;                 // A row (M = s)

  f32x8 acc[8] = {};
  for (int h = 0; h < NH; ++h) {
    const __bf16* Ab = AOT + (size_t)(b * NH + h) * NS * NE;
    const __bf16* Bb = weff + (size_t)(h * 4 + 3) * NE * NE;   // Wo_eff[h]
#pragma unroll
    for (int kk = 0; kk < 4; ++kk) {
      const int e0 = kk * 32;
      const __bf16* ap = Ab + (size_t)srow * NE + e0 + half * 8;
      bf16x16 av = ld2x8(ap, ap + 16);
#pragma unroll
      for (int n = 0; n < 8; ++n) {
        const __bf16* bp = Bb + (size_t)(n * 16 + l16) * NE + e0 + half * 8;
        bf16x16 bv = ld2x8(bp, bp + 16);
        acc[n] = wmma_bf16(av, bv, acc[n]);
      }
    }
  }
  // C [M=s][N=d] -> out[b][d][s]: lane has fixed d, 8 contiguous s
#pragma unroll
  for (int n = 0; n < 8; ++n) {
    int d = n * 16 + l16;
    int sbase = s0 + wave * 16 + half * 8;
    float* dp = out + ((size_t)b * NE + d) * NS + sbase;
    float4 lo = make_float4(acc[n][0], acc[n][1], acc[n][2], acc[n][3]);
    float4 hi = make_float4(acc[n][4], acc[n][5], acc[n][6], acc[n][7]);
    *(float4*)dp = lo;
    *(float4*)(dp + 4) = hi;
  }
}

extern "C" void kernel_launch(void* const* d_in, const int* in_sizes, int n_in,
                              void* d_out, int out_size, void* d_ws, size_t ws_size,
                              hipStream_t stream) {
  (void)in_sizes; (void)n_in; (void)out_size;
  if (ws_size < WS_NEED) return;  // need ~36 MB scratch

  const float* q  = (const float*)d_in[0];
  const float* k  = (const float*)d_in[1];
  const float* v  = (const float*)d_in[2];
  const float* Wq = (const float*)d_in[3];
  const float* Wk = (const float*)d_in[4];
  const float* Wv = (const float*)d_in[5];
  const float* Wo = (const float*)d_in[6];
  const float* lA = (const float*)d_in[7];
  const float* lB = (const float*)d_in[8];
  float* out = (float*)d_out;

  char* ws = (char*)d_ws;
  __bf16* weff = (__bf16*)(ws + OFF_WEFF);
  __bf16* qTi  = (__bf16*)(ws + OFF_QTI);
  __bf16* kTi  = (__bf16*)(ws + OFF_KTI);
  __bf16* vTi  = (__bf16*)(ws + OFF_VTI);
  __bf16* QT   = (__bf16*)(ws + OFF_QT);
  __bf16* KT   = (__bf16*)(ws + OFF_KT);
  __bf16* Vb   = (__bf16*)(ws + OFF_V);
  __bf16* AOT  = (__bf16*)(ws + OFF_AOT);

  k_merge<<<dim3((NH * 4 * NE * NE + 255) / 256), 256, 0, stream>>>(
      Wq, Wk, Wv, Wo, lA, lB, weff);
  k_tcvt<<<dim3((NB * NS * NE + 255) / 256, 1, 3), 256, 0, stream>>>(
      q, k, v, qTi, kTi, vTi);
  k_proj<<<dim3(NS / 64, NB * NH, 3), 128, 0, stream>>>(
      weff, qTi, kTi, vTi, QT, KT, Vb);
  k_attn<<<dim3(NS / 128, NB * NH), 128, (unsigned)SMEM_ATTN_B, stream>>>(
      QT, KT, Vb, AOT);
  k_out<<<dim3(NS / 64, NB), 128, 0, stream>>>(AOT, weff, out);
}
